// SGNS_5308579578065
// MI455X (gfx1250) — matrix operand order repaired and checked by reading, not compiled, not run on hardware
//
#include <hip/hip_runtime.h>
#include <hip/hip_bf16.h>

typedef __attribute__((ext_vector_type(16))) _Float16 v16h;
typedef __attribute__((ext_vector_type(8)))  float    v8f;

#define EMBED   64
#define CTX     20
#define NNEG    400        // CONTEXT * N_NEGS
#define BATCH   8192
#define WAVES_PER_BLOCK 8

// Numerically stable log-sigmoid: logsig(x) = min(x,0) - log(1 + exp(-|x|)).
// The log argument is in [1,2] (never denormal), so use the RAW hardware
// transcendentals (v_exp_f32 = 2^x, v_log_f32 = log2) without OCML's
// denorm-fixup wrappers:  min(x,0) - ln2 * log2(1 + 2^(-|x|*log2e)).
__device__ __forceinline__ float logsig(float x) {
    const float LOG2E = 1.44269504088896340736f;
    const float LN2   = 0.69314718055994530942f;
    float ax = __builtin_fabsf(x);
#if __has_builtin(__builtin_amdgcn_exp2f) && __has_builtin(__builtin_amdgcn_logf)
    float t = __builtin_amdgcn_exp2f(-ax * LOG2E);          // exp(-|x|)
    return fminf(x, 0.0f) - LN2 * __builtin_amdgcn_logf(1.0f + t);
#else
    return fminf(x, 0.0f) - __logf(1.0f + __expf(-ax));
#endif
}

// Build A fragment (16-bit A-matrix 16x32 layout, CDNA5 ISA 7.12.2):
//   lanes 0-15  (hi=0): halves 0..7 = K 0..7,  halves 8..15 = K 16..23
//   lanes 16-31 (hi=1): halves 0..7 = K 8..15, halves 8..15 = K 24..31
// K here maps to dim kblock+K of this lane's embedding row.
__device__ __forceinline__ v16h load_a_frag(const float* __restrict__ row, int kblock, int hi) {
    const int d0 = kblock + hi * 8;        // halves 0..7
    const int d1 = kblock + 16 + hi * 8;   // halves 8..15
    float4 x0 = *(const float4*)(row + d0);
    float4 x1 = *(const float4*)(row + d0 + 4);
    float4 y0 = *(const float4*)(row + d1);
    float4 y1 = *(const float4*)(row + d1 + 4);
    v16h a;
    a[0]  = (_Float16)x0.x; a[1]  = (_Float16)x0.y; a[2]  = (_Float16)x0.z; a[3]  = (_Float16)x0.w;
    a[4]  = (_Float16)x1.x; a[5]  = (_Float16)x1.y; a[6]  = (_Float16)x1.z; a[7]  = (_Float16)x1.w;
    a[8]  = (_Float16)y0.x; a[9]  = (_Float16)y0.y; a[10] = (_Float16)y0.z; a[11] = (_Float16)y0.w;
    a[12] = (_Float16)y1.x; a[13] = (_Float16)y1.y; a[14] = (_Float16)y1.z; a[15] = (_Float16)y1.w;
    return a;
}

// Build B fragment (16-bit B-matrix 32x16): same K order as A fragment:
// hi=0: halves 0..7 = K 0..7, 8..15 = K 16..23; hi=1: K 8..15, K 24..31;
// column index = lane%16. We replicate ivec
// into all 16 columns, so each lane's 16 halves depend only on hi.
__device__ __forceinline__ v16h load_b_frag(const float* __restrict__ iv, int kblock, int hi) {
    const int d0 = kblock + hi * 8;
    const int d1 = kblock + 16 + hi * 8;
    float4 x0 = *(const float4*)(iv + d0);
    float4 x1 = *(const float4*)(iv + d0 + 4);
    float4 x2 = *(const float4*)(iv + d1);
    float4 x3 = *(const float4*)(iv + d1 + 4);
    v16h b;
    b[0]  = (_Float16)x0.x; b[1]  = (_Float16)x0.y; b[2]  = (_Float16)x0.z; b[3]  = (_Float16)x0.w;
    b[4]  = (_Float16)x1.x; b[5]  = (_Float16)x1.y; b[6]  = (_Float16)x1.z; b[7]  = (_Float16)x1.w;
    b[8]  = (_Float16)x2.x; b[9]  = (_Float16)x2.y; b[10] = (_Float16)x2.z; b[11] = (_Float16)x2.w;
    b[12] = (_Float16)x3.x; b[13] = (_Float16)x3.y; b[14] = (_Float16)x3.z; b[15] = (_Float16)x3.w;
    return b;
}

__global__ void __launch_bounds__(256)
sgns_main(const int* __restrict__ iitem,
          const int* __restrict__ oitems,
          const int* __restrict__ nitems,
          const float* __restrict__ Wi,
          const float* __restrict__ Wo,
          float* __restrict__ partials) {
    const int tid  = threadIdx.x;
    const int wave = tid >> 5;               // 0..7
    const int lane = tid & 31;
    const int b    = blockIdx.x * WAVES_PER_BLOCK + wave;   // < 8192 always
    const int m    = lane & 15;              // A row within tile / B column
    const int hi   = lane >> 4;              // half-wave group

    // Center embedding -> B fragments, built once, reused for all 27 tiles.
    const float* iv = Wi + (size_t)iitem[b] * EMBED;
    const v16h b0 = load_b_frag(iv, 0,  hi);
    const v16h b1 = load_b_frag(iv, 32, hi);

    float partial = 0.0f;

    // ---- positives: 20 rows -> 2 tiles (tile 1 padded), logsig(+dot) ----
    const int* __restrict__ oidx = oitems + (size_t)b * CTX;
    #pragma unroll
    for (int t = 0; t < 2; ++t) {
        const int r   = t * 16 + m;
        const int idx = (r < CTX) ? oidx[r] : 0;          // clamp padding to row 0
        const float* row = Wo + (size_t)idx * EMBED;
        v8f acc = {};
        v16h a0 = load_a_frag(row, 0, hi);
        acc = __builtin_amdgcn_wmma_f32_16x16x32_f16(false, a0, false, b0,
                                                     (short)0, acc, false, false);
        v16h a1 = load_a_frag(row, 32, hi);
        acc = __builtin_amdgcn_wmma_f32_16x16x32_f16(false, a1, false, b1,
                                                     (short)0, acc, false, false);
        // D layout: lanes 0-15 VGPR v = row v; lanes 16-31 VGPR v = row v+8.
        // Results replicated across 16 columns; only column 0 lanes accumulate.
        if (m == 0) {
            #pragma unroll
            for (int v = 0; v < 8; ++v) {
                const int rr = t * 16 + hi * 8 + v;
                if (rr < CTX) partial += logsig(acc[v]);
            }
        }
    }

    // ---- negatives: 400 rows -> 25 full tiles, logsig(-dot) ----
    const int* __restrict__ nidx = nitems + (size_t)b * NNEG;
    for (int t = 0; t < 25; ++t) {
        const int idx = nidx[t * 16 + m];
        const float* row = Wo + (size_t)idx * EMBED;
        v8f acc = {};
        v16h a0 = load_a_frag(row, 0, hi);
        acc = __builtin_amdgcn_wmma_f32_16x16x32_f16(false, a0, false, b0,
                                                     (short)0, acc, false, false);
        v16h a1 = load_a_frag(row, 32, hi);
        acc = __builtin_amdgcn_wmma_f32_16x16x32_f16(false, a1, false, b1,
                                                     (short)0, acc, false, false);
        if (m == 0) {
            #pragma unroll
            for (int v = 0; v < 8; ++v) partial += logsig(-acc[v]);
        }
    }

    // ---- block reduction -> one partial per block (deterministic) ----
    __shared__ float sdata[256];
    sdata[tid] = partial;
    __syncthreads();
    #pragma unroll
    for (int s = 128; s > 0; s >>= 1) {
        if (tid < s) sdata[tid] += sdata[tid + s];
        __syncthreads();
    }
    if (tid == 0) partials[blockIdx.x] = sdata[0];
}

__global__ void __launch_bounds__(256)
sgns_reduce(const float* __restrict__ partials, int n, float* __restrict__ out) {
    __shared__ float sdata[256];
    float s = 0.0f;
    for (int i = threadIdx.x; i < n; i += 256) s += partials[i];
    sdata[threadIdx.x] = s;
    __syncthreads();
    #pragma unroll
    for (int k = 128; k > 0; k >>= 1) {
        if (threadIdx.x < k) sdata[threadIdx.x] += sdata[threadIdx.x + k];
        __syncthreads();
    }
    if (threadIdx.x == 0)
        out[0] = -sdata[0] / ((float)BATCH * (float)CTX);
}

extern "C" void kernel_launch(void* const* d_in, const int* in_sizes, int n_in,
                              void* d_out, int out_size, void* d_ws, size_t ws_size,
                              hipStream_t stream) {
    const int*   iitem  = (const int*)  d_in[0];   // [8192]
    const int*   oitems = (const int*)  d_in[1];   // [8192, 20]
    const int*   nitems = (const int*)  d_in[2];   // [8192, 400]
    const float* Wi     = (const float*)d_in[3];   // [1M, 64]
    const float* Wo     = (const float*)d_in[4];   // [1M, 64]
    float*       out    = (float*)      d_out;     // scalar
    float*       parts  = (float*)      d_ws;      // 1024 floats of scratch

    const int nblocks = BATCH / WAVES_PER_BLOCK;   // 1024
    sgns_main<<<nblocks, 256, 0, stream>>>(iitem, oitems, nitems, Wi, Wo, parts);
    sgns_reduce<<<1, 256, 0, stream>>>(parts, nblocks, out);
}
